// PointerGeneratorNetwork_18202071400480
// MI455X (gfx1250) — matrix operand order repaired and checked by reading, not compiled, not run on hardware
//
#include <hip/hip_runtime.h>

typedef __bf16 bf16;
typedef __attribute__((ext_vector_type(8)))  __bf16 v8bf;
typedef __attribute__((ext_vector_type(16))) __bf16 v16bf;
typedef __attribute__((ext_vector_type(8)))  float  v8f;

#define DEV static __device__ __forceinline__

DEV bf16 to_bf16(float f) {
  union { float f; unsigned u; } in; in.f = f;
  unsigned r = in.u + 0x7FFFu + ((in.u >> 16) & 1u);   // round-to-nearest-even
  union { unsigned short s; bf16 b; } out; out.s = (unsigned short)(r >> 16);
  return out.b;
}
DEV float bf16_to_f(bf16 b) {
  union { bf16 b; unsigned short s; } in; in.b = b;
  union { unsigned u; float f; } out; out.u = ((unsigned)in.s) << 16;
  return out.f;
}
DEV float sigmoidf(float x) { return 1.0f / (1.0f + __expf(-x)); }

union V16U { v16bf v; v8bf h[2]; };

// A fragment (16x32 bf16, row-major source, row = lane&15):
// lanes 0-15 hold K{0..7,16..23}; lanes 16-31 hold K{8..15,24..31}
DEV v16bf load_frag_a(const bf16* row, int k0, int lane) {
  int kh = (lane >> 4) & 1;
  V16U t;
  t.h[0] = *(const v8bf*)(row + k0 + kh * 8);
  t.h[1] = *(const v8bf*)(row + k0 + 16 + kh * 8);
  return t.v;
}
// B fragment (32x16 bf16; column n = lane&15 of B == row n of W for X@W^T):
// lanes 0-15 hold K{0..7,16..23}; lanes 16-31 hold K{8..15,24..31}
DEV v16bf load_frag_b(const bf16* row, int k0, int lane) {
  int kh = (lane >> 4) & 1;
  V16U t;
  t.h[0] = *(const v8bf*)(row + k0 + kh * 8);
  t.h[1] = *(const v8bf*)(row + k0 + 16 + kh * 8);
  return t.v;
}
DEV v8f wmma_bf16(v16bf a, v16bf b, v8f c) {
  return __builtin_amdgcn_wmma_f32_16x16x32_bf16(false, a, false, b, (short)0, c,
                                                 false, false);
}

// ---------------------------------------------------------------------------
// Generic GEMM: C[M,N] = A[M,K](bf16,row) @ W[N,K](bf16,row)^T + bias[N]
// one wave computes a 16(M) x 64(N) strip; K multiple of 32; M%16==0, N%64==0
// ---------------------------------------------------------------------------
__global__ void wmma_gemm_bf16(const bf16* __restrict__ A, const bf16* __restrict__ W,
                               const float* __restrict__ bias, float* __restrict__ C,
                               int M, int N, int K) {
  int wid  = (blockIdx.x * blockDim.x + threadIdx.x) >> 5;
  int lane = threadIdx.x & 31;
  int ngrp = N >> 6;
  int mt = wid / ngrp, ng = wid % ngrp;
  if (mt >= (M >> 4)) return;
  v8f acc0 = {}, acc1 = {}, acc2 = {}, acc3 = {};
  const bf16* arow  = A + (size_t)(mt * 16 + (lane & 15)) * K;
  const bf16* wbase = W + (size_t)(ng * 64 + (lane & 15)) * K;
  for (int k0 = 0; k0 < K; k0 += 32) {
    v16bf a = load_frag_a(arow, k0, lane);
    acc0 = wmma_bf16(a, load_frag_b(wbase + (size_t)0 * 16 * K, k0, lane), acc0);
    acc1 = wmma_bf16(a, load_frag_b(wbase + (size_t)1 * 16 * K, k0, lane), acc1);
    acc2 = wmma_bf16(a, load_frag_b(wbase + (size_t)2 * 16 * K, k0, lane), acc2);
    acc3 = wmma_bf16(a, load_frag_b(wbase + (size_t)3 * 16 * K, k0, lane), acc3);
  }
  int nl = lane & 15;
  int mb = mt * 16 + ((lane >> 4) << 3);
  v8f accs[4] = {acc0, acc1, acc2, acc3};
  for (int j = 0; j < 4; ++j) {
    int n = ng * 64 + j * 16 + nl;
    float bv = bias ? bias[n] : 0.0f;
    float* cp = C + (size_t)mb * N + n;
#pragma unroll
    for (int i = 0; i < 8; ++i) cp[(size_t)i * N] = accs[j][i] + bv;
  }
}

// ---------------------------------------------------------------------------
// Sequential LSTM over S steps, one workgroup (1024 thr = 32 waves).
// Gx[(b*S+t)*1024] holds X@Wih^T + bih + bhh. Wave w owns 16-col gate slice
// jt=w>>1 over batch half mt=w&1 -> i/f/g/o tiles live in its own accumulators.
// h (bf16) and c (f32) kept in LDS.
// ---------------------------------------------------------------------------
__global__ void lstm_seq_kernel(const float* __restrict__ Gx, const bf16* __restrict__ Whh,
                                bf16* __restrict__ seq_out, float* __restrict__ hc_final,
                                int S) {
  extern __shared__ char smem[];
  bf16*  h_lds = (bf16*)smem;               // 32*256 bf16 = 16 KB
  float* c_lds = (float*)(smem + 16384);    // 32*256 f32  = 32 KB
  const int tid = threadIdx.x, lane = tid & 31, w = tid >> 5;
  const int mt = w & 1, jt = w >> 1;
  for (int p = tid; p < 8192; p += 1024) { h_lds[p] = to_bf16(0.f); c_lds[p] = 0.f; }
  __syncthreads();
  const int nl = lane & 15;
  const int mb = mt * 16 + ((lane >> 4) << 3);
  const int j  = jt * 16 + nl;
  const bf16* arow = h_lds + (mt * 16 + nl) * 256;
  const bf16* w0 = Whh + (size_t)(0 * 256 + jt * 16 + nl) * 256;
  const bf16* w1 = Whh + (size_t)(1 * 256 + jt * 16 + nl) * 256;
  const bf16* w2 = Whh + (size_t)(2 * 256 + jt * 16 + nl) * 256;
  const bf16* w3 = Whh + (size_t)(3 * 256 + jt * 16 + nl) * 256;
  for (int t = 0; t < S; ++t) {
    v8f ai = {}, af = {}, ag = {}, ao = {};
    for (int k0 = 0; k0 < 256; k0 += 32) {
      v16bf a = load_frag_a(arow, k0, lane);
      ai = wmma_bf16(a, load_frag_b(w0, k0, lane), ai);
      af = wmma_bf16(a, load_frag_b(w1, k0, lane), af);
      ag = wmma_bf16(a, load_frag_b(w2, k0, lane), ag);
      ao = wmma_bf16(a, load_frag_b(w3, k0, lane), ao);
    }
    __syncthreads();   // all waves done reading old h
#pragma unroll
    for (int i = 0; i < 8; ++i) {
      int b = mb + i;
      const float* gx = Gx + ((size_t)b * S + t) * 1024;
      float iv = sigmoidf(ai[i] + gx[j]);
      float fv = sigmoidf(af[i] + gx[256 + j]);
      float gv = tanhf  (ag[i] + gx[512 + j]);
      float ov = sigmoidf(ao[i] + gx[768 + j]);
      float cn = fv * c_lds[b * 256 + j] + iv * gv;
      float hn = ov * tanhf(cn);
      c_lds[b * 256 + j] = cn;
      h_lds[b * 256 + j] = to_bf16(hn);
      if (seq_out) seq_out[((size_t)b * S + t) * 256 + j] = to_bf16(hn);
    }
    __syncthreads();
  }
  for (int p = tid; p < 8192; p += 1024) {
    hc_final[p]        = bf16_to_f(h_lds[p]);
    hc_final[8192 + p] = c_lds[p];
  }
}

// ---------------------------------------------------------------------------
// Decoder: T attention+LSTM steps, one workgroup. Emits h_t (bf16) and
// attention weights; generator GEMM/softmax/scatter are deferred.
// ---------------------------------------------------------------------------
__global__ void decoder_kernel(const float* __restrict__ Gdec,     // (B*T,1024) x-proj + biases
                               const bf16* __restrict__ Whh,       // dec_Whh (1024,256)
                               const bf16* __restrict__ Wc,        // dec_Wih[:,E:] (1024,256)
                               const bf16* __restrict__ Wah,       // attn_W[:,:H] (256,256)
                               const float* __restrict__ enc_proj, // (B*S,256) incl attn_b
                               const float* __restrict__ attn_v,   // (256)
                               const bf16* __restrict__ ctx_out,   // (B*S,256)
                               const float* __restrict__ hc_init,  // (2*32*256) qh,qc
                               bf16* __restrict__ hs_out,          // (B*T,256)
                               float* __restrict__ aw_out,         // (B*T,S)
                               int S, int T) {
  extern __shared__ char smem[];
  bf16*  h_lds  = (bf16*)smem;                // 16 KB
  bf16*  xc_lds = (bf16*)(smem + 16384);      // 16 KB
  float* c_lds  = (float*)(smem + 32768);     // 32 KB
  float* hWa    = (float*)(smem + 65536);     // 32 KB
  float* sc     = (float*)(smem + 98304);     // 32*400*4 = 50 KB
  float* av     = (float*)(smem + 149504);    // 1 KB
  const int tid = threadIdx.x, lane = tid & 31, w = tid >> 5;
  for (int p = tid; p < 8192; p += 1024) {
    h_lds[p] = to_bf16(hc_init[p]);
    c_lds[p] = hc_init[8192 + p];
  }
  for (int p = tid; p < 256; p += 1024) av[p] = attn_v[p];
  __syncthreads();
  const int nl = lane & 15;
  const int mt = w & 1, jt = w >> 1;
  const int mb = mt * 16 + ((lane >> 4) << 3);
  const int j  = jt * 16 + nl;
  const bf16* ah = h_lds  + (mt * 16 + nl) * 256;
  const bf16* ax = xc_lds + (mt * 16 + nl) * 256;
  const bf16* wh0 = Whh + (size_t)(0 * 256 + jt * 16 + nl) * 256;
  const bf16* wh1 = Whh + (size_t)(1 * 256 + jt * 16 + nl) * 256;
  const bf16* wh2 = Whh + (size_t)(2 * 256 + jt * 16 + nl) * 256;
  const bf16* wh3 = Whh + (size_t)(3 * 256 + jt * 16 + nl) * 256;
  const bf16* wc0 = Wc  + (size_t)(0 * 256 + jt * 16 + nl) * 256;
  const bf16* wc1 = Wc  + (size_t)(1 * 256 + jt * 16 + nl) * 256;
  const bf16* wc2 = Wc  + (size_t)(2 * 256 + jt * 16 + nl) * 256;
  const bf16* wc3 = Wc  + (size_t)(3 * 256 + jt * 16 + nl) * 256;
  for (int t = 0; t < T; ++t) {
    // 1. hWa = h @ Wa_h^T  (32x256); wave w -> tile (w&1, w>>1)
    {
      v8f acc = {};
      const bf16* wrow = Wah + (size_t)(jt * 16 + nl) * 256;
      for (int k0 = 0; k0 < 256; k0 += 32) {
        v16bf a = load_frag_a(ah, k0, lane);
        acc = wmma_bf16(a, load_frag_b(wrow, k0, lane), acc);
      }
#pragma unroll
      for (int i = 0; i < 8; ++i) hWa[(mb + i) * 256 + j] = acc[i];
    }
    __syncthreads();
    // 2. attention scores: e = tanh(hWa + enc_proj) . attn_v
    for (int p = tid; p < 32 * S; p += 1024) {
      int b = p / S, s = p - b * S;
      const float* ep = enc_proj + ((size_t)b * S + s) * 256;
      const float* hr = hWa + b * 256;
      float sum = 0.f;
      for (int hh = 0; hh < 256; ++hh) sum += tanhf(hr[hh] + ep[hh]) * av[hh];
      sc[p] = sum;
    }
    __syncthreads();
    // 3. softmax over S; wave w handles batch b=w
    {
      int b = w;
      float m = -3.4e38f;
      for (int s = lane; s < S; s += 32) m = fmaxf(m, sc[b * S + s]);
      for (int o = 16; o > 0; o >>= 1) m = fmaxf(m, __shfl_xor(m, o, 32));
      float ssum = 0.f;
      for (int s = lane; s < S; s += 32) {
        float e = __expf(sc[b * S + s] - m);
        sc[b * S + s] = e;
        ssum += e;
      }
      for (int o = 16; o > 0; o >>= 1) ssum += __shfl_xor(ssum, o, 32);
      float inv = 1.0f / ssum;
      for (int s = lane; s < S; s += 32) {
        float a = sc[b * S + s] * inv;
        sc[b * S + s] = a;
        aw_out[((size_t)b * T + t) * S + s] = a;
      }
    }
    __syncthreads();
    // 4. ctx_vec = aw @ ctx_out  -> xc_lds (bf16)
    for (int p = tid; p < 8192; p += 1024) {
      int b = p >> 8, jj = p & 255;
      float sum = 0.f;
      const bf16* cb = ctx_out + (size_t)b * S * 256 + jj;
      const float* ab = sc + b * S;
      for (int s = 0; s < S; ++s) sum += ab[s] * bf16_to_f(cb[(size_t)s * 256]);
      xc_lds[p] = to_bf16(sum);
    }
    __syncthreads();
    // 5. gates = Gdec[t] + h@Whh^T + ctx_vec@Wc^T
    v8f ai = {}, af = {}, ag = {}, ao = {};
    for (int k0 = 0; k0 < 256; k0 += 32) {
      v16bf a1 = load_frag_a(ah, k0, lane);
      v16bf a2 = load_frag_a(ax, k0, lane);
      ai = wmma_bf16(a1, load_frag_b(wh0, k0, lane), ai);
      ai = wmma_bf16(a2, load_frag_b(wc0, k0, lane), ai);
      af = wmma_bf16(a1, load_frag_b(wh1, k0, lane), af);
      af = wmma_bf16(a2, load_frag_b(wc1, k0, lane), af);
      ag = wmma_bf16(a1, load_frag_b(wh2, k0, lane), ag);
      ag = wmma_bf16(a2, load_frag_b(wc2, k0, lane), ag);
      ao = wmma_bf16(a1, load_frag_b(wh3, k0, lane), ao);
      ao = wmma_bf16(a2, load_frag_b(wc3, k0, lane), ao);
    }
    __syncthreads();
    // 6. LSTM update
#pragma unroll
    for (int i = 0; i < 8; ++i) {
      int b = mb + i;
      const float* gx = Gdec + ((size_t)b * T + t) * 1024;
      float iv = sigmoidf(ai[i] + gx[j]);
      float fv = sigmoidf(af[i] + gx[256 + j]);
      float gv = tanhf  (ag[i] + gx[512 + j]);
      float ov = sigmoidf(ao[i] + gx[768 + j]);
      float cn = fv * c_lds[b * 256 + j] + iv * gv;
      float hn = ov * tanhf(cn);
      c_lds[b * 256 + j] = cn;
      h_lds[b * 256 + j] = to_bf16(hn);
      hs_out[((size_t)b * T + t) * 256 + j] = to_bf16(hn);
    }
    __syncthreads();
  }
}

// ---------------------------------------------------------------------------
// Row softmax over V, in place (one block per row).
// ---------------------------------------------------------------------------
__global__ void softmax_rows_kernel(float* __restrict__ out, int V) {
  __shared__ float red[8];
  float* row = out + (size_t)blockIdx.x * V;
  int tid = threadIdx.x, lane = tid & 31, w = tid >> 5;
  float m = -3.4e38f;
  for (int v = tid; v < V; v += 256) m = fmaxf(m, row[v]);
  for (int o = 16; o > 0; o >>= 1) m = fmaxf(m, __shfl_xor(m, o, 32));
  if (lane == 0) red[w] = m;
  __syncthreads();
  float M = red[0];
  for (int i = 1; i < 8; ++i) M = fmaxf(M, red[i]);
  __syncthreads();
  float s = 0.f;
  for (int v = tid; v < V; v += 256) {
    float e = __expf(row[v] - M);
    row[v] = e;
    s += e;
  }
  for (int o = 16; o > 0; o >>= 1) s += __shfl_xor(s, o, 32);
  if (lane == 0) red[w] = s;
  __syncthreads();
  float S = 0.f;
  for (int i = 0; i < 8; ++i) S += red[i];
  float inv = 1.0f / S;
  for (int v = tid; v < V; v += 256) row[v] *= inv;
}

// out[b,t,context[b,s]] += aw[b,t,s] * (context != PAD)
__global__ void scatter_pointer_kernel(const float* __restrict__ aw,
                                       const int* __restrict__ context,
                                       float* __restrict__ out,
                                       int B, int T, int S, int V) {
  int p = blockIdx.x * blockDim.x + threadIdx.x;
  if (p >= B * T * S) return;
  int bt = p / S, s = p - bt * S;
  int b = bt / T;
  int tok = context[b * S + s];
  if (tok != 0)
    atomicAdd(out + (size_t)bt * V + tok, aw[(size_t)bt * S + s]);
}

// ------------------------------ prep kernels --------------------------------
__global__ void cast_f32_bf16_kernel(const float* __restrict__ src, bf16* __restrict__ dst,
                                     int rows, int cols, int src_ld, int src_off) {
  int p = blockIdx.x * blockDim.x + threadIdx.x;
  if (p >= rows * cols) return;
  int r = p / cols, c = p - r * cols;
  dst[p] = to_bf16(src[(size_t)r * src_ld + src_off + c]);
}
__global__ void add_bias_kernel(const float* __restrict__ a, const float* __restrict__ b,
                                float* __restrict__ o, int n) {
  int p = blockIdx.x * blockDim.x + threadIdx.x;
  if (p < n) o[p] = a[p] + b[p];
}
__global__ void gather_embed_kernel(const float* __restrict__ emb, const int* __restrict__ idx,
                                    bf16* __restrict__ out, int rows, int E) {
  int p = blockIdx.x * blockDim.x + threadIdx.x;
  if (p >= rows * E) return;
  int r = p / E, e = p - r * E;
  out[p] = to_bf16(emb[(size_t)idx[r] * E + e]);
}
__global__ void gather_dec_inputs_kernel(const float* __restrict__ emb,
                                         const int* __restrict__ answer,
                                         bf16* __restrict__ out, int B, int T, int E) {
  int p = blockIdx.x * blockDim.x + threadIdx.x;
  if (p >= B * T * E) return;
  int row = p / E, e = p - row * E;
  int b = row / T, t = row - b * T;
  int tok = (t == 0) ? 1 : answer[b * T + t - 1];   // START = 1
  out[p] = to_bf16(emb[(size_t)tok * E + e]);
}

// ---------------------------------------------------------------------------
extern "C" void kernel_launch(void* const* d_in, const int* in_sizes, int n_in,
                              void* d_out, int out_size, void* d_ws, size_t ws_size,
                              hipStream_t stream) {
  (void)in_sizes; (void)n_in; (void)out_size; (void)ws_size;
  constexpr int V = 32000, E = 256, H = 256, B = 32, S = 400, SQ = 32, T = 32, FH = 1024;
  const float* emb_enc = (const float*)d_in[0];
  const float* emb_dec = (const float*)d_in[1];
  const float* enc_Wih = (const float*)d_in[2];
  const float* enc_Whh = (const float*)d_in[3];
  const float* enc_bih = (const float*)d_in[4];
  const float* enc_bhh = (const float*)d_in[5];
  const float* dec_Wih = (const float*)d_in[6];
  const float* dec_Whh = (const float*)d_in[7];
  const float* dec_bih = (const float*)d_in[8];
  const float* dec_bhh = (const float*)d_in[9];
  const float* attn_W  = (const float*)d_in[10];
  const float* attn_b  = (const float*)d_in[11];
  const float* attn_v  = (const float*)d_in[12];
  const float* gen_W   = (const float*)d_in[13];
  const float* gen_b   = (const float*)d_in[14];
  const int* context   = (const int*)d_in[15];
  const int* question  = (const int*)d_in[16];
  const int* answer    = (const int*)d_in[17];

  char* base = (char*)d_ws;
  size_t off = 0;
  auto carve = [&](size_t bytes) -> void* {
    void* p = base + off;
    off = (off + bytes + 255) & ~(size_t)255;
    return p;
  };
  bf16*  Xctx    = (bf16*)carve((size_t)B * S * E * 2);
  bf16*  Xq      = (bf16*)carve((size_t)B * SQ * E * 2);
  bf16*  Xdec    = (bf16*)carve((size_t)B * T * E * 2);
  bf16*  wEncIh  = (bf16*)carve((size_t)FH * E * 2);
  bf16*  wEncHh  = (bf16*)carve((size_t)FH * H * 2);
  bf16*  wDecX   = (bf16*)carve((size_t)FH * E * 2);
  bf16*  wDecC   = (bf16*)carve((size_t)FH * H * 2);
  bf16*  wDecHh  = (bf16*)carve((size_t)FH * H * 2);
  bf16*  wWah    = (bf16*)carve((size_t)H * H * 2);
  bf16*  wWae    = (bf16*)carve((size_t)H * H * 2);
  bf16*  wGen    = (bf16*)carve((size_t)V * H * 2);
  float* encBias = (float*)carve((size_t)FH * 4);
  float* decBias = (float*)carve((size_t)FH * 4);
  float* Gctx    = (float*)carve((size_t)B * S * FH * 4);
  float* Gq      = (float*)carve((size_t)B * SQ * FH * 4);
  float* Gdec    = (float*)carve((size_t)B * T * FH * 4);
  bf16*  ctxOut  = (bf16*)carve((size_t)B * S * H * 2);
  float* encProj = (float*)carve((size_t)B * S * H * 4);
  float* qhc     = (float*)carve((size_t)2 * B * H * 4);
  float* dhc     = (float*)carve((size_t)2 * B * H * 4);
  bf16*  Hs      = (bf16*)carve((size_t)B * T * H * 2);
  float* Aw      = (float*)carve((size_t)B * T * S * 4);

  auto g1 = [](int n) { return (n + 255) / 256; };
  auto gg = [](int M, int N) { return ((M / 16) * (N / 64) * 32 + 255) / 256; };

  // weight casts / bias folds / embedding gathers
  cast_f32_bf16_kernel<<<g1(FH * E), 256, 0, stream>>>(enc_Wih, wEncIh, FH, E, E, 0);
  cast_f32_bf16_kernel<<<g1(FH * H), 256, 0, stream>>>(enc_Whh, wEncHh, FH, H, H, 0);
  cast_f32_bf16_kernel<<<g1(FH * E), 256, 0, stream>>>(dec_Wih, wDecX, FH, E, E + H, 0);
  cast_f32_bf16_kernel<<<g1(FH * H), 256, 0, stream>>>(dec_Wih, wDecC, FH, H, E + H, E);
  cast_f32_bf16_kernel<<<g1(FH * H), 256, 0, stream>>>(dec_Whh, wDecHh, FH, H, H, 0);
  cast_f32_bf16_kernel<<<g1(H * H), 256, 0, stream>>>(attn_W, wWah, H, H, 2 * H, 0);
  cast_f32_bf16_kernel<<<g1(H * H), 256, 0, stream>>>(attn_W, wWae, H, H, 2 * H, H);
  cast_f32_bf16_kernel<<<g1(V * H), 256, 0, stream>>>(gen_W, wGen, V, H, H, 0);
  add_bias_kernel<<<g1(FH), 256, 0, stream>>>(enc_bih, enc_bhh, encBias, FH);
  add_bias_kernel<<<g1(FH), 256, 0, stream>>>(dec_bih, dec_bhh, decBias, FH);
  gather_embed_kernel<<<g1(B * S * E), 256, 0, stream>>>(emb_enc, context, Xctx, B * S, E);
  gather_embed_kernel<<<g1(B * SQ * E), 256, 0, stream>>>(emb_enc, question, Xq, B * SQ, E);
  gather_dec_inputs_kernel<<<g1(B * T * E), 256, 0, stream>>>(emb_dec, answer, Xdec, B, T, E);

  // batched input projections (time-parallel)
  wmma_gemm_bf16<<<gg(B * S, FH), 256, 0, stream>>>(Xctx, wEncIh, encBias, Gctx, B * S, FH, E);
  wmma_gemm_bf16<<<gg(B * SQ, FH), 256, 0, stream>>>(Xq, wEncIh, encBias, Gq, B * SQ, FH, E);
  wmma_gemm_bf16<<<gg(B * T, FH), 256, 0, stream>>>(Xdec, wDecX, decBias, Gdec, B * T, FH, E);

  // sequential encoders (h,c resident in LDS)
  lstm_seq_kernel<<<1, 1024, 49152, stream>>>(Gctx, wEncHh, ctxOut, dhc, S);
  lstm_seq_kernel<<<1, 1024, 49152, stream>>>(Gq, wEncHh, (bf16*)nullptr, qhc, SQ);

  // enc_proj = ctx_out @ Wa_e^T + attn_b (once, not per decode step)
  wmma_gemm_bf16<<<gg(B * S, H), 256, 0, stream>>>(ctxOut, wWae, attn_b, encProj, B * S, H, H);

  // sequential decoder with attention
  decoder_kernel<<<1, 1024, 150528, stream>>>(Gdec, wDecHh, wDecC, wWah, encProj, attn_v,
                                              ctxOut, qhc, Hs, Aw, S, T);

  // generator: logits straight into d_out, then softmax + pointer scatter
  wmma_gemm_bf16<<<gg(B * T, V), 256, 0, stream>>>(Hs, wGen, gen_b, (float*)d_out, B * T, V, H);
  softmax_rows_kernel<<<B * T, 256, 0, stream>>>((float*)d_out, V);
  scatter_pointer_kernel<<<g1(B * T * S), 256, 0, stream>>>(Aw, context, (float*)d_out,
                                                            B, T, S, V);
}
